// XmmLayer_22634477650637
// MI455X (gfx1250) — matrix operand order repaired and checked by reading, not compiled, not run on hardware
//
#include <hip/hip_runtime.h>

// ---------------------------------------------------------------------------
// Fused polynomial GEMM + BatchNorm for MI455X (gfx1250, wave32, WMMA + TDM).
//   y = x@c1^T + (x^2)@c2^T + (x^3)@c3^T + rowsum(bias)   (B=4096, OUT=IN=1024)
//   out = (y - mean_b(y)) * rsqrt(var_b(y)+eps) * gamma + beta
//
// Single K=3072 GEMM on A=[x|x^2|x^3], B=[c1|c2|c3] using
// v_wmma_f32_16x16x32_bf16 with hi/lo bf16 operand split (hi*hi+hi*lo+lo*hi).
// A/B tiles are staged in LDS by the Tensor Data Mover (tensor_load_to_lds),
// double-buffered, synchronized with s_wait_tensorcnt + workgroup barrier.
// ---------------------------------------------------------------------------

typedef __attribute__((ext_vector_type(16))) __bf16        v16bf;
typedef __attribute__((ext_vector_type(8)))  float         v8f;
typedef __attribute__((ext_vector_type(4)))  unsigned int  v4u;
typedef __attribute__((ext_vector_type(8)))  int           v8i;
typedef __attribute__((ext_vector_type(4)))  int           v4i;

#define B_DIM   4096
#define OUT_DIM 1024
#define IN_DIM  1024
#define K3      (3 * IN_DIM)     // 3072
#define BN_EPS  1e-5f

// LDS tile geometry: 128 rows x 32 bf16 (64B) per row, padded to 80B rows
// (TDM pad: every 16 DWORDs insert 4 DWORDs) -> conflict-free ds_load_b128.
#define ROW_ELEMS   40                       // 80 bytes / 2
#define REGION_ELEM (128 * ROW_ELEMS)        // 5120 bf16 elements
#define REGION_BYTE (REGION_ELEM * 2)        // 10240 bytes
#define BUF_BYTE    (4 * REGION_BYTE)        // A_hi, A_lo, B_hi, B_lo
#define LDS_BYTES   (2 * BUF_BYTE)           // double buffered: 80 KB

// ---- bf16 helpers (round-to-nearest-even, manual to guarantee codegen) ----
__device__ __forceinline__ unsigned short f2bf(float f) {
  unsigned u = __float_as_uint(f);
  u += 0x7FFFu + ((u >> 16) & 1u);
  return (unsigned short)(u >> 16);
}
__device__ __forceinline__ float bf2f(unsigned short h) {
  return __uint_as_float(((unsigned)h) << 16);
}
__device__ __forceinline__ void split2(float v, unsigned short& hi, unsigned short& lo) {
  hi = f2bf(v);
  lo = f2bf(v - bf2f(hi));
}

// ---------------------------------------------------------------------------
// Pack A = [x | x^2 | x^3]  (B_DIM x K3) as hi/lo bf16.
// ---------------------------------------------------------------------------
__global__ void pack_x_kernel(const float* __restrict__ x,
                              unsigned short* __restrict__ ahi,
                              unsigned short* __restrict__ alo) {
  int idx = blockIdx.x * blockDim.x + threadIdx.x;   // 0 .. B*IN-1
  int b = idx / IN_DIM;
  int i = idx - b * IN_DIM;
  float v  = x[idx];
  float v2 = v * v;
  float v3 = v2 * v;
  size_t base = (size_t)b * K3;
  unsigned short h, l;
  split2(v,  h, l); ahi[base + i]              = h; alo[base + i]              = l;
  split2(v2, h, l); ahi[base + IN_DIM + i]     = h; alo[base + IN_DIM + i]     = l;
  split2(v3, h, l); ahi[base + 2 * IN_DIM + i] = h; alo[base + 2 * IN_DIM + i] = l;
}

// ---------------------------------------------------------------------------
// Pack B = [c1 | c2 | c3]  (OUT_DIM x K3, row-major over OUT) as hi/lo bf16.
// ---------------------------------------------------------------------------
__global__ void pack_c_kernel(const float* __restrict__ c1,
                              const float* __restrict__ c2,
                              const float* __restrict__ c3,
                              unsigned short* __restrict__ bhi,
                              unsigned short* __restrict__ blo) {
  int idx = blockIdx.x * blockDim.x + threadIdx.x;   // 0 .. OUT*IN-1
  int n = idx / IN_DIM;
  int k = idx - n * IN_DIM;
  size_t base = (size_t)n * K3;
  unsigned short h, l;
  split2(c1[idx], h, l); bhi[base + k]              = h; blo[base + k]              = l;
  split2(c2[idx], h, l); bhi[base + IN_DIM + k]     = h; blo[base + IN_DIM + k]     = l;
  split2(c3[idx], h, l); bhi[base + 2 * IN_DIM + k] = h; blo[base + 2 * IN_DIM + k] = l;
}

// ---------------------------------------------------------------------------
// rowsum(bias): one block per output row, LDS tree reduction.
// ---------------------------------------------------------------------------
__global__ void bias_rowsum_kernel(const float* __restrict__ bias,
                                   float* __restrict__ bsum) {
  __shared__ float red[256];
  int o = blockIdx.x;
  float s = 0.f;
  for (int i = threadIdx.x; i < IN_DIM; i += 256) s += bias[(size_t)o * IN_DIM + i];
  red[threadIdx.x] = s;
  __syncthreads();
  for (int off = 128; off > 0; off >>= 1) {
    if (threadIdx.x < (unsigned)off) red[threadIdx.x] += red[threadIdx.x + off];
    __syncthreads();
  }
  if (threadIdx.x == 0) bsum[o] = red[0];
}

// ---------------------------------------------------------------------------
// TDM: issue one 128x32-bf16 tile load (global -> LDS with 16B/row padding).
// D# fields per CDNA5 ISA sec. 8.3-8.6. All inputs are wave-uniform.
// This toolchain exposes the 6-arg builtin:
//   (uint32x4 g0, int32x8 g1, int32x4 g2, int32x4 g3, int32x8 g4, i32 cpol)
// Groups 2/3/4 are zero: 2D tensor, no gather/iterate/multicast state.
// ---------------------------------------------------------------------------
__device__ __forceinline__ void tdm_issue_tile(const unsigned short* gptr,
                                               unsigned ldsOff,
                                               unsigned tensorRows) {
  unsigned long long ga = (unsigned long long)(uintptr_t)gptr;
  v4u g0;
  g0[0] = 1u;                                        // count=1 (valid), user mode
  g0[1] = ldsOff;                                    // lds_addr (bytes)
  g0[2] = (unsigned)ga;                              // global_addr[31:0]
  g0[3] = ((unsigned)(ga >> 32) & 0x01FFFFFFu)       // global_addr[56:32]
          | (2u << 30);                              // type=2 ("image")
  v8i g1;
  g1[0] = (1 << 16)                                  // data_size = 2 bytes
        | (1 << 20)                                  // pad_enable
        | (3 << 22)                                  // pad_interval: 16 DWORDs (64B)
        | (3 << 25);                                 // pad_amount:   4 DWORDs (16B)
  g1[1] = (int)((K3 & 0xFFFFu) << 16);               // tensor_dim0[15:0]
  g1[2] = (int)(((unsigned)K3 >> 16)                 // tensor_dim0[31:16]
        | ((tensorRows & 0xFFFFu) << 16));           // tensor_dim1[15:0]
  g1[3] = (int)((tensorRows >> 16)                   // tensor_dim1[31:16]
        | (32u << 16));                              // tile_dim0 = 32 elements
  g1[4] = 128;                                       // tile_dim1 = 128, tile_dim2 = 0
  g1[5] = K3;                                        // tensor_dim0_stride[31:0]
  g1[6] = 0;                                         // stride hi / dim1_stride lo
  g1[7] = 0;
  v4i z4 = {0, 0, 0, 0};                             // groups 2/3 unused (2D)
  v8i z8 = {0, 0, 0, 0, 0, 0, 0, 0};
  __builtin_amdgcn_tensor_load_to_lds(g0, g1, z4, z4, z8, 0);
}

// ---------------------------------------------------------------------------
// WMMA GEMM: 4 waves/block; each wave computes a 64x64 tile (4x4 of 16x16);
// block covers 128x128. Per K-step each wave TDM-loads one of the four
// 128x32 tile regions (A_hi/A_lo/B_hi/B_lo), double-buffered in LDS.
// ---------------------------------------------------------------------------
union Frag {
  uint4 u[2];
  v16bf v;
};

__launch_bounds__(128)
__global__ void gemm_wmma_kernel(const unsigned short* __restrict__ Ah,
                                 const unsigned short* __restrict__ Al,
                                 const unsigned short* __restrict__ Bh,
                                 const unsigned short* __restrict__ Bl,
                                 const float* __restrict__ bsum,
                                 float* __restrict__ Y) {
  extern __shared__ unsigned short lds[];            // LDS_BYTES dynamic

  const int wave = threadIdx.x >> 5;
  const int lane = threadIdx.x & 31;
  const int half = lane >> 4;                        // 0: lanes 0-15, 1: 16-31
  const int r    = lane & 15;

  const int mBlock = blockIdx.y * 128;               // block-level row base
  const int nBlock = blockIdx.x * 128;               // block-level col base
  const int aRow0  = (wave >> 1) * 64;               // wave offsets inside block
  const int bRow0  = (wave & 1) * 64;

  const unsigned ldsBase = (unsigned)(uintptr_t)(void*)lds;

  // Per-wave TDM region: wave0->A_hi, wave1->A_lo, wave2->B_hi, wave3->B_lo.
  const unsigned short* gsrc;
  unsigned tRows;
  int gRow0;
  if (wave < 2) { gsrc = (wave == 0) ? Ah : Al; tRows = B_DIM;   gRow0 = mBlock; }
  else          { gsrc = (wave == 2) ? Bh : Bl; tRows = OUT_DIM; gRow0 = nBlock; }

  // Prologue: fill buffer 0.
  tdm_issue_tile(gsrc + (size_t)gRow0 * K3, ldsBase + (unsigned)wave * REGION_BYTE, tRows);
  __builtin_amdgcn_s_wait_tensorcnt(0);
  __syncthreads();

  v8f acc[4][4] = {};

  for (int k0 = 0; k0 < K3; k0 += 32) {
    const int buf = (k0 >> 5) & 1;
    const bool haveNext = (k0 + 32) < K3;

    // Kick off next buffer's tile (overlaps with compute below).
    if (haveNext) {
      tdm_issue_tile(gsrc + (size_t)gRow0 * K3 + (k0 + 32),
                     ldsBase + (unsigned)(buf ^ 1) * BUF_BYTE +
                         (unsigned)wave * REGION_BYTE,
                     tRows);
    }

    const unsigned short* bb  = lds + (size_t)buf * (BUF_BYTE / 2);
    const unsigned short* aHi = bb;
    const unsigned short* aLo = bb + REGION_ELEM;
    const unsigned short* bHi = bb + 2 * REGION_ELEM;
    const unsigned short* bLo = bb + 3 * REGION_ELEM;

    Frag afh[4], afl[4], bfh[4], bfl[4];
    // A fragment gather (contiguous K, matching B):
    //   lane<16:  VGPR0-7 = K[0..15]
    //   lane>=16: VGPR0-7 = K[16..31]
#pragma unroll
    for (int t = 0; t < 4; ++t) {
      int rowA = aRow0 + t * 16 + r;
      const uint4* ph = (const uint4*)(aHi + rowA * ROW_ELEMS);
      const uint4* pl = (const uint4*)(aLo + rowA * ROW_ELEMS);
      afh[t].u[0] = ph[2 * half];  afh[t].u[1] = ph[2 * half + 1];
      afl[t].u[0] = pl[2 * half];  afl[t].u[1] = pl[2 * half + 1];
      // B fragment: column n = r, 16 contiguous K values per lane-half.
      int rowB = bRow0 + t * 16 + r;
      const uint4* qh = (const uint4*)(bHi + rowB * ROW_ELEMS + half * 16);
      const uint4* ql = (const uint4*)(bLo + rowB * ROW_ELEMS + half * 16);
      bfh[t].u[0] = qh[0];  bfh[t].u[1] = qh[1];
      bfl[t].u[0] = ql[0];  bfl[t].u[1] = ql[1];
    }

#pragma unroll
    for (int mt = 0; mt < 4; ++mt) {
#pragma unroll
      for (int nt = 0; nt < 4; ++nt) {
        acc[mt][nt] = __builtin_amdgcn_wmma_f32_16x16x32_bf16(
            false, afh[mt].v, false, bfh[nt].v, (short)0, acc[mt][nt], false, false);
        acc[mt][nt] = __builtin_amdgcn_wmma_f32_16x16x32_bf16(
            false, afh[mt].v, false, bfl[nt].v, (short)0, acc[mt][nt], false, false);
        acc[mt][nt] = __builtin_amdgcn_wmma_f32_16x16x32_bf16(
            false, afl[mt].v, false, bfh[nt].v, (short)0, acc[mt][nt], false, false);
      }
    }

    // Next buffer complete (own wave's TDM), then all waves rendezvous:
    // barrier also guarantees everyone is done reading `buf` before it is
    // overwritten in the following iteration.
    if (haveNext) __builtin_amdgcn_s_wait_tensorcnt(0);
    __syncthreads();
  }

  // Epilogue: C/D layout -> VGPR j holds row m = 8*half + j, col n = r.
#pragma unroll
  for (int nt = 0; nt < 4; ++nt) {
    int n = nBlock + bRow0 + nt * 16 + r;
    float bs = bsum[n];
#pragma unroll
    for (int mt = 0; mt < 4; ++mt) {
#pragma unroll
      for (int j = 0; j < 8; ++j) {
        int m = mBlock + aRow0 + mt * 16 + half * 8 + j;
        Y[(size_t)m * OUT_DIM + n] = acc[mt][nt][j] + bs;
      }
    }
  }
}

// ---------------------------------------------------------------------------
// BatchNorm stats: deterministic two-stage column reduction (no float atomics).
// ---------------------------------------------------------------------------
__global__ void colstats_partial_kernel(const float* __restrict__ Y,
                                        float* __restrict__ psum,
                                        float* __restrict__ psumsq) {
  int col = blockIdx.x * 256 + threadIdx.x;   // grid.x = 4
  int rc  = blockIdx.y;                       // grid.y = 8
  float s = 0.f, ss = 0.f;
  int b0 = rc * (B_DIM / 8);
  for (int b = b0; b < b0 + (B_DIM / 8); ++b) {
    float v = Y[(size_t)b * OUT_DIM + col];
    s += v;
    ss += v * v;
  }
  psum[rc * OUT_DIM + col]   = s;
  psumsq[rc * OUT_DIM + col] = ss;
}

__global__ void colstats_final_kernel(const float* __restrict__ psum,
                                      const float* __restrict__ psumsq,
                                      const float* __restrict__ gamma,
                                      const float* __restrict__ beta,
                                      float* __restrict__ scale,
                                      float* __restrict__ shift) {
  int col = blockIdx.x * 256 + threadIdx.x;   // grid.x = 4
  float s = 0.f, ss = 0.f;
  for (int rc = 0; rc < 8; ++rc) {
    s  += psum[rc * OUT_DIM + col];
    ss += psumsq[rc * OUT_DIM + col];
  }
  float mean = s * (1.f / B_DIM);
  float var  = ss * (1.f / B_DIM) - mean * mean;
  float rstd = rsqrtf(var + BN_EPS);
  float g = gamma[col] * rstd;
  scale[col] = g;
  shift[col] = beta[col] - mean * g;
}

__global__ void bn_apply_kernel(float* __restrict__ Y,
                                const float* __restrict__ scale,
                                const float* __restrict__ shift) {
  size_t idx = (size_t)blockIdx.x * blockDim.x + threadIdx.x;
  int col = (int)(idx & (OUT_DIM - 1));
  Y[idx] = Y[idx] * scale[col] + shift[col];
}

// ---------------------------------------------------------------------------
// Host-side launch. All kernels on `stream`; workspace carved from d_ws.
// ---------------------------------------------------------------------------
extern "C" void kernel_launch(void* const* d_in, const int* in_sizes, int n_in,
                              void* d_out, int out_size, void* d_ws, size_t ws_size,
                              hipStream_t stream) {
  (void)in_sizes; (void)n_in; (void)out_size; (void)ws_size;

  const float* x     = (const float*)d_in[0];
  const float* c1    = (const float*)d_in[1];
  const float* c2    = (const float*)d_in[2];
  const float* c3    = (const float*)d_in[3];
  const float* bias  = (const float*)d_in[4];
  const float* gamma = (const float*)d_in[5];
  const float* beta  = (const float*)d_in[6];
  float* Y = (float*)d_out;                    // y lives in d_out (4096x1024 f32)

  // Workspace layout.
  char* ws = (char*)d_ws;
  const size_t szA = (size_t)B_DIM * K3 * sizeof(unsigned short);    // 24 MB
  const size_t szB = (size_t)OUT_DIM * K3 * sizeof(unsigned short);  //  6 MB
  unsigned short* Ah = (unsigned short*)ws;              ws += szA;
  unsigned short* Al = (unsigned short*)ws;              ws += szA;
  unsigned short* Bh = (unsigned short*)ws;              ws += szB;
  unsigned short* Bl = (unsigned short*)ws;              ws += szB;
  float* bsum   = (float*)ws;                            ws += OUT_DIM * sizeof(float);
  float* psum   = (float*)ws;                            ws += 8 * OUT_DIM * sizeof(float);
  float* psumsq = (float*)ws;                            ws += 8 * OUT_DIM * sizeof(float);
  float* scale  = (float*)ws;                            ws += OUT_DIM * sizeof(float);
  float* shift  = (float*)ws;

  // 1) Pack operands to split-bf16.
  pack_x_kernel<<<(B_DIM * IN_DIM) / 256, 256, 0, stream>>>(x, Ah, Al);
  pack_c_kernel<<<(OUT_DIM * IN_DIM) / 256, 256, 0, stream>>>(c1, c2, c3, Bh, Bl);

  // 2) Bias row sums.
  bias_rowsum_kernel<<<OUT_DIM, 256, 0, stream>>>(bias, bsum);

  // 3) Fused K=3072 WMMA GEMM with TDM-staged LDS tiles (+bias) -> Y.
  dim3 grid(OUT_DIM / 128, B_DIM / 128);   // (8, 32)
  gemm_wmma_kernel<<<grid, 128, LDS_BYTES, stream>>>(Ah, Al, Bh, Bl, bsum, Y);

  // 4) BatchNorm (training-mode batch stats), in place on Y.
  colstats_partial_kernel<<<dim3(OUT_DIM / 256, 8), 256, 0, stream>>>(Y, psum, psumsq);
  colstats_final_kernel<<<OUT_DIM / 256, 256, 0, stream>>>(psum, psumsq, gamma, beta, scale, shift);
  bn_apply_kernel<<<(size_t)B_DIM * OUT_DIM / 256, 256, 0, stream>>>(Y, scale, shift);
}
